// TimescaleLoss_52364241273576
// MI455X (gfx1250) — compile-verified, matrix operand order checked by reading, not run on hardware
//
#include <hip/hip_runtime.h>

typedef __attribute__((ext_vector_type(16))) _Float16 v16h;
typedef __attribute__((ext_vector_type(8)))  float    v8f;

#define NB   32      // batch
#define NL   256     // channels
#define NHW  256     // spatial (16*16)
#define IMG  (NL * NHW)
#define BINSZ 20.0f

// |xj - xi| for two elements, packed as two f16 in one dword.
// f16 is sign-magnitude, so abs == clear sign bits (exact). This avoids the
// gfx1250 ISel crash on f32-fabs-of-packed-fadd patterns.
__device__ __forceinline__ unsigned absdiff_pk(float xj0, float xi0,
                                               float xj1, float xi1) {
    union { _Float16 h[2]; unsigned u; } t;
    t.h[0] = (_Float16)(xj0 - xi0);
    t.h[1] = (_Float16)(xj1 - xi1);
    return t.u & 0x7FFF7FFFu;
}

// One block per (i,j) pair (upper triangle does work). 256 threads = 8 waves.
// Wave w handles channel groups {2w, 2w+1} of 16 channels each.
// Spatial L1 reduction on the matrix unit: C += |Xj-Xi| (16x32 f16) * Ones(32x16).
__global__ __launch_bounds__(256)
void timescale_pair_kernel(const float* __restrict__ latents,
                           const int*   __restrict__ ts,
                           float*       __restrict__ dist)
{
    const int pair = blockIdx.x;
    const int i = pair >> 5;
    const int j = pair & 31;
    const int tid = threadIdx.x;

    if (i > j) {                       // symmetry: reducer doubles i<j terms
        if (tid == 0) dist[pair] = 0.0f;
        return;
    }

    // bins = clip(ceil(log2(|tj-ti|+1)*20), 0, 256)  (uniform across block)
    const int   dti = abs(ts[j] - ts[i]);
    const float dt  = (float)dti + 1.0f;
    int bins = (int)ceilf(log2f(dt) * BINSZ);
    bins = max(0, min(bins, NL));

    const float* Xi = latents + (size_t)i * IMG;
    const float* Xj = latents + (size_t)j * IMG;

    const int wave = tid >> 5;
    const int lane = tid & 31;
    const int m    = lane & 15;   // row within 16-channel group
    const int half = lane >> 4;   // K-half select per WMMA A-layout

    v16h ones;
    #pragma unroll
    for (int e = 0; e < 16; ++e) ones[e] = (_Float16)1.0f;

    float acc = 0.0f;

    #pragma unroll
    for (int gsub = 0; gsub < 2; ++gsub) {
        const int c0 = (wave * 2 + gsub) * 16;
        const float* pi = Xi + (c0 + m) * NHW;
        const float* pj = Xj + (c0 + m) * NHW;

        v8f C = {};
        #pragma unroll
        for (int s = 0; s < NHW; s += 32) {
            // 16-bit A 16x32 layout: lanes 0-15 hold K = s+[0..7], s+[16..23];
            //                        lanes 16-31 hold K = s+[8..15], s+[24..31].
            const int o1 = s + 8 * half;
            const int o2 = s + 16 + 8 * half;
            float4 ia = *(const float4*)(pi + o1);
            float4 ib = *(const float4*)(pi + o1 + 4);
            float4 ic = *(const float4*)(pi + o2);
            float4 id = *(const float4*)(pi + o2 + 4);
            float4 ja = *(const float4*)(pj + o1);
            float4 jb = *(const float4*)(pj + o1 + 4);
            float4 jc = *(const float4*)(pj + o2);
            float4 jd = *(const float4*)(pj + o2 + 4);

            union { v16h h; unsigned u[8]; } a;
            a.u[0] = absdiff_pk(ja.x, ia.x, ja.y, ia.y);
            a.u[1] = absdiff_pk(ja.z, ia.z, ja.w, ia.w);
            a.u[2] = absdiff_pk(jb.x, ib.x, jb.y, ib.y);
            a.u[3] = absdiff_pk(jb.z, ib.z, jb.w, ib.w);
            a.u[4] = absdiff_pk(jc.x, ic.x, jc.y, ic.y);
            a.u[5] = absdiff_pk(jc.z, ic.z, jc.w, ic.w);
            a.u[6] = absdiff_pk(jd.x, id.x, jd.y, id.y);
            a.u[7] = absdiff_pk(jd.z, id.z, jd.w, id.w);

            // C[m,n] += sum_k A[m,k] * 1  -> per-channel spatial L1 in f32
            C = __builtin_amdgcn_wmma_f32_16x16x32_f16(
                    false, a.h, false, ones, (short)0, C, false, false);
        }

        // C VGPR r on this lane = D[c0 + r + 8*half] (all columns identical).
        // Apply suffix mask (ch >= bins) and weight w = 2^((255-ch)/20).
        #pragma unroll
        for (int r = 0; r < 8; ++r) {
            const int ch = c0 + r + 8 * half;
            const float w = exp2f((float)(255 - ch) * 0.05f);
            acc += (ch >= bins) ? w * C[r] : 0.0f;
        }
    }

    // Wave combine: lanes 0-15 carry identical half=0 sums, lanes 16-31 half=1.
    float tot = __shfl(acc, 0, 32) + __shfl(acc, 16, 32);

    __shared__ float wsum[8];
    if (lane == 0) wsum[wave] = tot;
    __syncthreads();
    if (tid == 0) {
        float s = 0.0f;
        #pragma unroll
        for (int w = 0; w < 8; ++w) s += wsum[w];
        dist[pair] = s;
    }
}

// Deterministic symmetric mean over the 32x32 pair matrix.
__global__ __launch_bounds__(256)
void timescale_reduce_kernel(const float* __restrict__ dist,
                             float*       __restrict__ out)
{
    __shared__ float s[256];
    const int t = threadIdx.x;
    float v = 0.0f;
    for (int p = t; p < NB * NB; p += 256) {
        const int i = p >> 5, j = p & 31;
        const float w = (i < j) ? 2.0f : ((i == j) ? 1.0f : 0.0f);
        v += w * dist[p];
    }
    s[t] = v;
    __syncthreads();
    for (int off = 128; off > 0; off >>= 1) {
        if (t < off) s[t] += s[t + off];
        __syncthreads();
    }
    if (t == 0) out[0] = s[0] * (1.0f / (float)(NB * NB));
}

extern "C" void kernel_launch(void* const* d_in, const int* in_sizes, int n_in,
                              void* d_out, int out_size, void* d_ws, size_t ws_size,
                              hipStream_t stream) {
    const float* latents = (const float*)d_in[0];   // [32, 256, 16, 16] f32
    const int*   tsteps  = (const int*)d_in[1];     // [32] i32
    float* out  = (float*)d_out;                    // scalar f32
    float* dist = (float*)d_ws;                     // 1024 f32 pair distances

    timescale_pair_kernel<<<NB * NB, 256, 0, stream>>>(latents, tsteps, dist);
    timescale_reduce_kernel<<<1, 256, 0, stream>>>(dist, out);
}
